// MinimalAtlasBlock_60722247630991
// MI455X (gfx1250) — matrix operand order repaired and checked by reading, not compiled, not run on hardware
//
#include <hip/hip_runtime.h>
#include <hip/hip_bf16.h>
#include <math.h>

// ---------------------------------------------------------------------------
// MinimalAtlasBlock for MI455X (gfx1250).
// B=16, L=512, D=128.  Dominant cost: per-step Newton-Schulz X*X^T*X
// (two 128^3 matmuls, sequential over t) -> one workgroup per batch,
// M/S state resident in LDS (320KB/WGP), bf16 WMMA for the O(D^3) chain.
// K-outer / 4-accumulator tiling: each wave owns a 16x64 output band,
// loads its A fragment once per K-slice and feeds 4 WMMAs with it.
// ---------------------------------------------------------------------------

#define Bsz 16
#define Lsz 512
#define Dsz 128
#define PM  129   // f32 LDS pitch (bank-conflict-free column walks)
#define PB  136   // bf16 LDS pitch (rows stay 16B aligned: 136*2=272)

typedef unsigned short u16t;
typedef __attribute__((ext_vector_type(8)))  u16t   ush8;
typedef __attribute__((ext_vector_type(16))) u16t   ush16;
typedef __attribute__((ext_vector_type(16))) __bf16 v16bf;
typedef __attribute__((ext_vector_type(8)))  float  v8f;

// LDS offsets (bytes) inside dynamic smem
#define OFF_M   0
#define OFF_S   (OFF_M  + Dsz*PM*4)       // 66048
#define OFF_X   (OFF_S  + Dsz*PM*4)       // 132096
#define OFF_XT  (OFF_X  + Dsz*PB*2)       // 166912
#define OFF_A   (OFF_XT + Dsz*PB*2)       // 201728
#define OFF_KV  (OFF_A  + Dsz*PB*2)       // 236544
#define OFF_QV  (OFF_KV + Dsz*4)
#define OFF_XV  (OFF_QV + Dsz*4)
#define OFF_ERR (OFF_XV + Dsz*4)
#define OFF_RED (OFF_ERR+ Dsz*4)
#define OFF_INV (OFF_RED+ 16*4)
#define SMEM_BYTES (OFF_INV + 64)

__device__ __forceinline__ float wave_sum(float v) {
    #pragma unroll
    for (int o = 16; o > 0; o >>= 1) v += __shfl_xor(v, o, 32);
    return v;
}

// Build a 16xbf16 fragment from two contiguous 16B LDS chunks.
__device__ __forceinline__ v16bf load_frag(const __bf16* p0, const __bf16* p1) {
    ush8 a = *(const ush8*)p0;
    ush8 b = *(const ush8*)p1;
    ush16 w = __builtin_shufflevector(a, b, 0,1,2,3,4,5,6,7,8,9,10,11,12,13,14,15);
    return __builtin_bit_cast(v16bf, w);
}

// A-operand fragment: 16x32 bf16 tile, rows r0..r0+15, K k0..k0+31, from
// row-major src (pitch PB).  ISA layout: lane half h reads K {8h..8h+7} and
// {16+8h..16+8h+7} of row (r0 + lane&15), each an aligned 16B chunk.
__device__ __forceinline__ v16bf load_a_frag(const __bf16* src, int r0, int k0, int lane) {
    const int m = lane & 15, h = lane >> 4;
    const __bf16* row = src + (r0 + m) * PB;
    return load_frag(row + k0 + 8*h, row + k0 + 16 + 8*h);
}

// B-operand fragment: 32x16 bf16 tile B[k][n] with B^T given row-major
// (pitch PB).  Lane half h needs K {8h..8h+7} and {16+8h..16+8h+7} of column
// (c0 + lane&15) = two aligned 16B chunks of row (c0+n) of B^T (matches A).
__device__ __forceinline__ v16bf load_b_frag(const __bf16* bt, int c0, int k0, int lane) {
    const int n = lane & 15, h = lane >> 4;
    const __bf16* row = bt + (c0 + n) * PB + k0;
    return load_frag(row + 8*h, row + 16 + 8*h);
}

// ---------------------------------------------------------------------------
// Kernel 1: per-token projections.  block = one (b,l) row, 128 threads.
//   q = x*Wq^T ; k = normalize(x*Wk^T) ; qa = tanh(exp(lg)*(q @ P0))
//   k_phi = c0*k + c1*k^2 ; q_phi = c0*qa + c1*qa^2
// ---------------------------------------------------------------------------
__global__ __launch_bounds__(128) void atlas_proj_kernel(
    const float* __restrict__ x, const float* __restrict__ Wq,
    const float* __restrict__ Wk, const float* __restrict__ P0,
    const float* __restrict__ log_gain, const float* __restrict__ coeffs,
    float* __restrict__ kphi, float* __restrict__ qphi)
{
    __shared__ float sx[Dsz], sq[Dsz], swred[4];
    const int tid = threadIdx.x;
    const int row = blockIdx.x;            // b*L + l
    const float* xr = x + (size_t)row * Dsz;
    sx[tid] = xr[tid];
    __syncthreads();

    const float c0 = coeffs[0], c1 = coeffs[1];
    float qe = 0.f, ke = 0.f;
    const float* wqr = Wq + tid * Dsz;
    const float* wkr = Wk + tid * Dsz;
    #pragma unroll 8
    for (int d = 0; d < Dsz; ++d) { qe = fmaf(sx[d], wqr[d], qe); ke = fmaf(sx[d], wkr[d], ke); }
    sq[tid] = qe;

    float p = wave_sum(ke * ke);
    if ((tid & 31) == 0) swred[tid >> 5] = p;
    __syncthreads();
    const float nrm = sqrtf(swred[0] + swred[1] + swred[2] + swred[3]);
    const float kn = ke / fmaxf(nrm, 1e-12f);
    kphi[(size_t)row * Dsz + tid] = c0 * kn + c1 * kn * kn;

    float ge = 0.f;                         // (q @ P0)[e] = sum_d q[d]*P0[d,e]
    #pragma unroll 8
    for (int d = 0; d < Dsz; ++d) ge = fmaf(sq[d], P0[d * Dsz + tid], ge);
    const float qa = tanhf(expf(log_gain[tid]) * ge);
    qphi[(size_t)row * Dsz + tid] = c0 * qa + c1 * qa * qa;
}

// ---------------------------------------------------------------------------
// Kernel 2: sequential scan.  One block per batch; M,S in LDS; per step:
//   err = M k - x ; S = .9 S + err k^T ; X = S/||S||F (bf16, + transpose)
//   A = X X^T (wmma) ; G = A X (wmma) ; M = .99M - .015 X + .005 G ; y = M q
// ---------------------------------------------------------------------------
__global__ __launch_bounds__(512) void atlas_scan_kernel(
    const float* __restrict__ kphi, const float* __restrict__ qphi,
    const float* __restrict__ x, const float* __restrict__ M0,
    const float* __restrict__ S0, float* __restrict__ y)
{
    extern __shared__ char smem[];
    float*  sM   = (float*)(smem + OFF_M);
    float*  sS   = (float*)(smem + OFF_S);
    __bf16* sX   = (__bf16*)(smem + OFF_X);
    __bf16* sXt  = (__bf16*)(smem + OFF_XT);
    __bf16* sA   = (__bf16*)(smem + OFF_A);
    float*  kv   = (float*)(smem + OFF_KV);
    float*  qv   = (float*)(smem + OFF_QV);
    float*  xv   = (float*)(smem + OFF_XV);
    float*  errv = (float*)(smem + OFF_ERR);
    float*  red  = (float*)(smem + OFF_RED);
    float*  sInv = (float*)(smem + OFF_INV);

    const int tid  = threadIdx.x;
    const int lane = tid & 31;
    const int wv   = tid >> 5;              // 16 waves
    const int b    = blockIdx.x;

    // Each wave owns a 16x64 output band: rows ti*16.., cols tj0*16..+63.
    const int ti  = wv >> 1;                // 0..7
    const int tj0 = (wv & 1) * 4;           // 0 or 4
    const int m   = lane & 15, h = lane >> 4;

    // init per-batch state (M0/S0 broadcast over b)
    #pragma unroll
    for (int e = 0; e < 32; ++e) {
        const int lin = tid * 32 + e, i = lin >> 7, j = lin & 127;
        sM[i * PM + j] = M0[lin];
        sS[i * PM + j] = S0[lin];
    }
    __syncthreads();

    const size_t bbase = (size_t)b * Lsz * Dsz;

    for (int t = 0; t < Lsz; ++t) {
        const size_t rb = bbase + (size_t)t * Dsz;
        // --- load step vectors + prefetch next step ---
        if (tid < Dsz) { kv[tid] = kphi[rb + tid]; qv[tid] = qphi[rb + tid]; xv[tid] = x[rb + tid]; }
        if (t + 1 < Lsz && tid < 2) {
            const size_t nb = rb + Dsz + (size_t)tid * 64;
            __builtin_prefetch(kphi + nb, 0, 0);
            __builtin_prefetch(qphi + nb, 0, 0);
            __builtin_prefetch(x    + nb, 0, 0);
        }
        __syncthreads();

        // --- err = M k - x ---
        if (tid < Dsz) {
            const float* mr = sM + tid * PM;
            float acc = 0.f;
            #pragma unroll 8
            for (int j = 0; j < Dsz; ++j) acc = fmaf(mr[j], kv[j], acc);
            errv[tid] = acc - xv[tid];
        }
        __syncthreads();

        // --- S = .9 S + err k^T ; accumulate ||S||F^2 ---
        float loc = 0.f;
        #pragma unroll
        for (int e = 0; e < 32; ++e) {
            const int lin = tid * 32 + e, i = lin >> 7, j = lin & 127;
            const int idx = i * PM + j;
            const float s = 0.9f * sS[idx] + errv[i] * kv[j];
            sS[idx] = s;
            loc = fmaf(s, s, loc);
        }
        loc = wave_sum(loc);
        if (lane == 0) red[wv] = loc;
        __syncthreads();
        if (tid == 0) {
            float tot = 0.f;
            #pragma unroll
            for (int i = 0; i < 16; ++i) tot += red[i];
            *sInv = 1.f / (sqrtf(tot) + 1e-6f);
        }
        __syncthreads();
        const float inv = *sInv;

        // --- X = S*inv (bf16) and X^T ---
        #pragma unroll
        for (int e = 0; e < 32; ++e) {
            const int lin = tid * 32 + e, i = lin >> 7, j = lin & 127;
            const float v = sS[i * PM + j] * inv;
            sX [i * PB + j] = (__bf16)v;
            sXt[j * PB + i] = (__bf16)v;
        }
        __syncthreads();

        // --- A = X * X^T  (B^T = X, row-major in LDS) ---
        {
            v8f acc[4] = {{}, {}, {}, {}};
            #pragma unroll
            for (int kk = 0; kk < 4; ++kk) {
                const int k0 = kk * 32;
                const v16bf a = load_a_frag(sX, ti * 16, k0, lane);   // loaded once
                #pragma unroll
                for (int j = 0; j < 4; ++j) {
                    const v16bf bf = load_b_frag(sX, (tj0 + j) * 16, k0, lane);
                    acc[j] = __builtin_amdgcn_wmma_f32_16x16x32_bf16(false, a, false, bf,
                                                                    (short)0, acc[j], false, false);
                }
            }
            #pragma unroll
            for (int j = 0; j < 4; ++j)
                #pragma unroll
                for (int r = 0; r < 8; ++r)
                    sA[(ti * 16 + r + 8 * h) * PB + (tj0 + j) * 16 + m] = (__bf16)acc[j][r];
        }
        __syncthreads();

        // --- G = A * X  (B^T = X^T) ; fused M = .99M - .015*inv*S + .005G ---
        {
            v8f acc[4] = {{}, {}, {}, {}};
            #pragma unroll
            for (int kk = 0; kk < 4; ++kk) {
                const int k0 = kk * 32;
                const v16bf a = load_a_frag(sA, ti * 16, k0, lane);   // loaded once
                #pragma unroll
                for (int j = 0; j < 4; ++j) {
                    const v16bf bf = load_b_frag(sXt, (tj0 + j) * 16, k0, lane);
                    acc[j] = __builtin_amdgcn_wmma_f32_16x16x32_bf16(false, a, false, bf,
                                                                    (short)0, acc[j], false, false);
                }
            }
            #pragma unroll
            for (int j = 0; j < 4; ++j)
                #pragma unroll
                for (int r = 0; r < 8; ++r) {
                    const int idx = (ti * 16 + r + 8 * h) * PM + (tj0 + j) * 16 + m;
                    sM[idx] = 0.99f * sM[idx] - 0.015f * inv * sS[idx] + 0.005f * acc[j][r];
                }
        }
        __syncthreads();

        // --- y_t = M q ---
        if (tid < Dsz) {
            const float* mr = sM + tid * PM;
            float acc = 0.f;
            #pragma unroll 8
            for (int j = 0; j < Dsz; ++j) acc = fmaf(mr[j], qv[j], acc);
            y[rb + tid] = acc;
        }
        __syncthreads();
    }
}

// ---------------------------------------------------------------------------
// Kernel 3: out = y * Wout^T + bout.  block = one (b,l) row, 128 threads.
// ---------------------------------------------------------------------------
__global__ __launch_bounds__(128) void atlas_out_kernel(
    const float* __restrict__ y, const float* __restrict__ Wout,
    const float* __restrict__ bout, float* __restrict__ out)
{
    __shared__ float sy[Dsz];
    const int tid = threadIdx.x;
    const int row = blockIdx.x;
    sy[tid] = y[(size_t)row * Dsz + tid];
    __syncthreads();
    const float* wr = Wout + tid * Dsz;
    float acc = bout[tid];
    #pragma unroll 8
    for (int d = 0; d < Dsz; ++d) acc = fmaf(sy[d], wr[d], acc);
    out[(size_t)row * Dsz + tid] = acc;
}

// ---------------------------------------------------------------------------
extern "C" void kernel_launch(void* const* d_in, const int* in_sizes, int n_in,
                              void* d_out, int out_size, void* d_ws, size_t ws_size,
                              hipStream_t stream) {
    const float* x        = (const float*)d_in[0];
    const float* Wq       = (const float*)d_in[1];
    const float* Wk       = (const float*)d_in[2];
    const float* P0       = (const float*)d_in[3];
    const float* M0       = (const float*)d_in[4];
    const float* S0       = (const float*)d_in[5];
    const float* log_gain = (const float*)d_in[6];
    const float* coeffs   = (const float*)d_in[7];
    const float* Wout     = (const float*)d_in[8];
    const float* bout     = (const float*)d_in[9];
    float* out = (float*)d_out;

    const size_t N = (size_t)Bsz * Lsz * Dsz;      // 1,048,576 elements
    float* kphi = (float*)d_ws;
    float* qphi = kphi + N;
    float* yv   = qphi + N;

    (void)in_sizes; (void)n_in; (void)out_size; (void)ws_size;

    hipFuncSetAttribute((const void*)atlas_scan_kernel,
                        hipFuncAttributeMaxDynamicSharedMemorySize, SMEM_BYTES);

    atlas_proj_kernel<<<Bsz * Lsz, Dsz, 0, stream>>>(x, Wq, Wk, P0, log_gain,
                                                     coeffs, kphi, qphi);
    atlas_scan_kernel<<<Bsz, 512, SMEM_BYTES, stream>>>(kphi, qphi, x, M0, S0, yv);
    atlas_out_kernel<<<Bsz * Lsz, Dsz, 0, stream>>>(yv, Wout, bout, out);
}
